// Model_63075889709681
// MI455X (gfx1250) — compile-verified, matrix operand order checked by reading, not run on hardware
//
#include <hip/hip_runtime.h>

typedef __attribute__((ext_vector_type(8)))  _Float16 v8h;
typedef __attribute__((ext_vector_type(16))) _Float16 v16h;
typedef __attribute__((ext_vector_type(8)))  float    v8f;

#define HD    10000   // hypervector dimensions
#define HDP   10048   // padded row stride (multiple of 64)
#define NPIX  784     // 28*28 pixels
#define NB    16      // batch
#define NLEV  1000    // levels
#define NCLS  10      // classes
#define PCH   112     // pixels per bundle block (7 chunks * 112 = 784)
#define DTH   128     // threads (d-lanes) per bundle block

// ---------------------------------------------------------------------------
// Kernel 1: 2D grid (k, b). Zero accumulator, build zero-padded f16 weight
// matrix WH[16][HDP] (classes>=10 and k>=HD are 0), and compute level indices.
// ---------------------------------------------------------------------------
__global__ void hdc_prep(const float* __restrict__ x,     // [16, 784]
                         const float* __restrict__ W,     // [10, HD]
                         float*    __restrict__ acc,      // [16, HDP]
                         _Float16* __restrict__ WH,       // [16, HDP]
                         int*      __restrict__ idx) {    // [16, 784]
    const int k = blockIdx.x * blockDim.x + threadIdx.x;
    const int b = blockIdx.y;
    if (k >= HDP) return;

    acc[b * HDP + k] = 0.0f;

    float w = 0.0f;
    if (b < NCLS && k < HD) w = W[b * HD + k];
    WH[b * HDP + k] = (_Float16)w;

    if (k < NPIX) {
        float v = rintf(x[b * NPIX + k] * (float)(NLEV - 1)); // nearest-even = jnp.round
        int i = (int)v;
        i = i < 0 ? 0 : (i > NLEV - 1 ? NLEV - 1 : i);
        idx[b * NPIX + k] = i;
    }
}

// ---------------------------------------------------------------------------
// Kernel 2: bundling  acc[b,d] += sum_p value[idx[b,p], d] * pos[p, d]
// Block owns 128 consecutive d's and a 112-pixel chunk; 16*112 indices staged
// in LDS so gather row bases are wave-uniform (scalar-offset + lane d).
// ---------------------------------------------------------------------------
__global__ void hdc_bundle(const float* __restrict__ pos,   // [784, HD]
                           const float* __restrict__ val,   // [1000, HD]
                           const int*   __restrict__ idx,   // [16, 784]
                           float*       __restrict__ acc) { // [16, HDP]
    __shared__ int sidx[PCH * NB];

    const int tid = threadIdx.x;
    const int p0  = blockIdx.y * PCH;

    for (int i = tid; i < PCH * NB; i += DTH) {
        const int pl = i >> 4;        // pixel within chunk
        const int b  = i & 15;        // batch
        sidx[i] = idx[b * NPIX + p0 + pl];
    }
    __syncthreads();

    const int  d     = blockIdx.x * DTH + tid;
    const bool valid = d < HD;
    const int  dd    = valid ? d : 0;

    float a[NB];
#pragma unroll
    for (int b = 0; b < NB; ++b) a[b] = 0.0f;

    for (int pl = 0; pl < PCH; ++pl) {
        const int p = p0 + pl;
        const float pv = pos[(size_t)p * HD + dd];
        if (pl + 1 < PCH)
            __builtin_prefetch(&pos[(size_t)(p + 1) * HD + dd], 0, 1);
        const int* row = &sidx[pl * NB];
#pragma unroll
        for (int b = 0; b < NB; ++b) {
            const float vv = val[(size_t)row[b] * HD + dd];
            a[b] = fmaf(vv, pv, a[b]);
        }
    }

    if (valid) {
#pragma unroll
        for (int b = 0; b < NB; ++b)
            atomicAdd(&acc[b * HDP + d], a[b]);
    }
}

// ---------------------------------------------------------------------------
// Kernel 3: hard-quantize accumulator into zero-padded f16 enc matrix.
// encH[b,k] = k<HD ? (acc>0 ? +1 : -1) : 0
// ---------------------------------------------------------------------------
__global__ void hdc_quant(const float* __restrict__ acc,
                          _Float16*    __restrict__ encH) {
    const int k = blockIdx.x * blockDim.x + threadIdx.x;
    const int b = blockIdx.y;
    if (k >= HDP) return;
    float e = 0.0f;
    if (k < HD) e = (acc[b * HDP + k] > 0.0f) ? 1.0f : -1.0f;
    encH[b * HDP + k] = (_Float16)e;
}

// ---------------------------------------------------------------------------
// Kernel 4: out[16,10] = enc @ W^T via f16 WMMA, one wave32.
// Per ISA 7.12.2, a lane's 16-bit A/B fragment = two contiguous 8-half runs:
//   K in [k0 + 8*hi, +8) and [k0 + 16 + 8*hi, +8), lane%16 = M (A) / N (B).
// So each fragment is exactly two 16-byte loads from the padded row.
// Two independent accumulators break the WMMA D->C dependency chain.
// ---------------------------------------------------------------------------
__global__ void hdc_classify(const _Float16* __restrict__ encH, // [16, HDP]
                             const _Float16* __restrict__ WH,   // [16, HDP]
                             float*          __restrict__ out) {// [16, 10]
    const int lane = threadIdx.x;   // 0..31
    const int row  = lane & 15;     // batch for A, class for B
    const int hi   = lane >> 4;     // 0/1 half selector

    const _Float16* ap = encH + row * HDP + 8 * hi;
    const _Float16* bp = WH   + row * HDP + 8 * hi;

    v8f c0 = {}, c1 = {};
    for (int k0 = 0; k0 < HDP; k0 += 64) {   // 157 iterations, 2 WMMA each
        __builtin_prefetch(ap + k0 + 64, 0, 1);
        __builtin_prefetch(bp + k0 + 64, 0, 1);

        v8h al0 = *(const v8h*)(ap + k0);
        v8h ah0 = *(const v8h*)(ap + k0 + 16);
        v8h bl0 = *(const v8h*)(bp + k0);
        v8h bh0 = *(const v8h*)(bp + k0 + 16);
        v16h a0 = __builtin_shufflevector(al0, ah0, 0,1,2,3,4,5,6,7,8,9,10,11,12,13,14,15);
        v16h b0 = __builtin_shufflevector(bl0, bh0, 0,1,2,3,4,5,6,7,8,9,10,11,12,13,14,15);
        c0 = __builtin_amdgcn_wmma_f32_16x16x32_f16(false, a0, false, b0,
                                                    (short)0, c0, false, false);

        v8h al1 = *(const v8h*)(ap + k0 + 32);
        v8h ah1 = *(const v8h*)(ap + k0 + 48);
        v8h bl1 = *(const v8h*)(bp + k0 + 32);
        v8h bh1 = *(const v8h*)(bp + k0 + 48);
        v16h a1 = __builtin_shufflevector(al1, ah1, 0,1,2,3,4,5,6,7,8,9,10,11,12,13,14,15);
        v16h b1 = __builtin_shufflevector(bl1, bh1, 0,1,2,3,4,5,6,7,8,9,10,11,12,13,14,15);
        c1 = __builtin_amdgcn_wmma_f32_16x16x32_f16(false, a1, false, b1,
                                                    (short)0, c1, false, false);
    }

#pragma unroll
    for (int r = 0; r < 8; ++r) {
        const int m = r + 8 * hi;   // batch (C/D layout: VGPR r -> M = r + 8*(lane/16))
        const int n = row;          // class (lane%16 = N)
        if (n < NCLS) out[m * NCLS + n] = c0[r] + c1[r];
    }
}

// ---------------------------------------------------------------------------
// ws layout:
//   [0)                acc  : 16*HDP f32  = 643,072 B
//   [643072)           idx  : 16*784 int  =  50,176 B
//   [693248)           encH : 16*HDP f16  = 321,536 B
//   [1014784)          WH   : 16*HDP f16  = 321,536 B   (total ~1.34 MB)
// ---------------------------------------------------------------------------
extern "C" void kernel_launch(void* const* d_in, const int* in_sizes, int n_in,
                              void* d_out, int out_size, void* d_ws, size_t ws_size,
                              hipStream_t stream) {
    const float* x   = (const float*)d_in[0];  // [16,28,28]
    const float* pos = (const float*)d_in[1];  // [784, HD]
    const float* val = (const float*)d_in[2];  // [1000, HD]
    const float* W   = (const float*)d_in[3];  // [10, HD]
    float* out = (float*)d_out;                // [16, 10]

    char* ws = (char*)d_ws;
    float*    acc  = (float*)ws;
    int*      idx  = (int*)(ws + 643072);
    _Float16* encH = (_Float16*)(ws + 693248);
    _Float16* WH   = (_Float16*)(ws + 1014784);

    dim3 g2d((HDP + 255) / 256, NB);           // 40 x 16
    hdc_prep<<<g2d, 256, 0, stream>>>(x, W, acc, WH, idx);

    dim3 gb((HD + DTH - 1) / DTH, NPIX / PCH); // 79 x 7
    hdc_bundle<<<gb, DTH, 0, stream>>>(pos, val, idx, acc);

    hdc_quant<<<g2d, 256, 0, stream>>>(acc, encH);

    hdc_classify<<<1, 32, 0, stream>>>(encH, WH, out);
}
